// Layer_7078106104191
// MI455X (gfx1250) — matrix operand order repaired and checked by reading, not compiled, not run on hardware
//
#include <hip/hip_runtime.h>
#include <math.h>

#define N_NODES 20000
#define N_EDGES 320000

typedef __attribute__((ext_vector_type(16))) __bf16 v16bf;
typedef __attribute__((ext_vector_type(8)))  float  v8f;

// ---------------------------------------------------------------------------
// WMMA helpers (CDNA5 wave32, v_wmma_f32_16x16x32_bf16)
// A fragment: lane l -> row m = l&15, khalf = l>>4.
//   elem j in [0,8):  K = k0 + khalf*8 + j
//   elem j in [8,16): K = k0 + 16 + khalf*8 + (j-8)
// B fragment: lane l -> col n = (l&15)+n0, khalf = l>>4; elem j -> K as for A
// C/D:        lane l -> col n = l&15, vgpr v -> row m = v + 8*(l>>4)
// ---------------------------------------------------------------------------

__device__ inline v8f wmma_bf16(v16bf a, v16bf b, v8f c) {
  return __builtin_amdgcn_wmma_f32_16x16x32_bf16(
      /*neg_a=*/false, a, /*neg_b=*/false, b,
      /*c_mod=*/(short)0, c, /*reuse_a=*/false, /*reuse_b=*/false);
}

// Load a 16x32 A-tile from row-major fp32 X (row stride ld, element stride st)
__device__ inline v16bf load_a_frag(const float* X, int ld, int st,
                                    int row0, int k0, int lane) {
  const int m  = lane & 15;
  const int kh = (lane >> 4) & 1;
  const float* row = X + (size_t)(row0 + m) * ld;
  const int kb = k0 + kh * 8;
  v16bf a;
#pragma unroll
  for (int j = 0; j < 8; ++j)  a[j]     = (__bf16)row[(kb + j) * st];
#pragma unroll
  for (int j = 0; j < 8; ++j)  a[8 + j] = (__bf16)row[(kb + 16 + j) * st];
  return a;
}

// Load a 32x16 B-tile from a row-major fp32 weight W (m rows x 32 cols)
__device__ inline v16bf load_b_frag(const float* W, int k0, int n0, int lane) {
  const int n  = (lane & 15) + n0;
  const int kh = (lane >> 4) & 1;
  v16bf b;
#pragma unroll
  for (int j = 0; j < 16; ++j) b[j] = (__bf16)W[(size_t)(k0 + kh * 8 + (j & 7) + (j >> 3) * 16) * 32 + n];
  return b;
}

// Store C fragment (16x16) into a 16x32 fp32 LDS tile at column offset n0
__device__ inline void store_c_lds(float* tile, v8f c, int n0, int lane) {
  const int n  = (lane & 15) + n0;
  const int mb = (lane >> 4) * 8;
#pragma unroll
  for (int v = 0; v < 8; ++v) tile[(mb + v) * 32 + n] = c[v];
}

// Store C fragment to out: out[(row0+m)*384 + colbase + n*colstride]
__device__ inline void store_c_out(float* out, int row0, int colbase,
                                   int colstride, v8f c, int n0, int lane) {
  const int n  = (lane & 15) + n0;
  const int mb = (lane >> 4) * 8;
#pragma unroll
  for (int v = 0; v < 8; ++v)
    out[(size_t)(row0 + mb + v) * 384 + colbase + n * colstride] = c[v];
}

// Branch-free tanh-approx GELU (jax.nn.gelu approximate=True).
// tanh(u) = sign(u) * (1 - e) / (1 + e),  e = exp(-2|u|)  (stable, e in (0,1])
__device__ inline float gelu_tanh(float x) {
  const float c0 = 0.7978845608028654f;   // sqrt(2/pi)
  float u  = c0 * (x + 0.044715f * x * x * x);
  float e  = __expf(-2.0f * fabsf(u));
  float t  = (1.0f - e) / (1.0f + e);
  t = copysignf(t, u);
  return 0.5f * x * (1.0f + t);
}

// ---------------------------------------------------------------------------
// Edge kernel: one wave (32 lanes) per edge, lane = channel k.
// Computes l=0/1/2 tensor-product messages and scatter-adds into fp32
// accumulators (resident in L2) with hardware global_atomic_add_f32.
// ---------------------------------------------------------------------------
__global__ __launch_bounds__(256) void edge_kernel(
    const float* __restrict__ pos, const float* __restrict__ x0,
    const float* __restrict__ x1, const int* __restrict__ senders,
    const int* __restrict__ receivers, float* __restrict__ A0,
    float* __restrict__ A1o, float* __restrict__ A1e, float* __restrict__ A2e) {
  const int e = blockIdx.x * 8 + (threadIdx.x >> 5);
  if (e >= N_EDGES) return;
  const int k = threadIdx.x & 31;
  const int s = senders[e];
  const int r = receivers[e];

  const float vx = pos[(size_t)r * 3 + 0] - pos[(size_t)s * 3 + 0];
  const float vy = pos[(size_t)r * 3 + 1] - pos[(size_t)s * 3 + 1];
  const float vz = pos[(size_t)r * 3 + 2] - pos[(size_t)s * 3 + 2];
  float rn = sqrtf(vx * vx + vy * vy + vz * vz);
  rn = fmaxf(rn, 1e-12f);
  const float sc = 1.7320508075688772f / rn;   // sqrt(3)/r
  const float bx = vx * sc, by = vy * sc, bz = vz * sc;

  const float s0 = x0[(size_t)s * 32 + k];
  const float* s1p = x1 + ((size_t)s * 32 + k) * 3;
  const float ax = s1p[0], ay = s1p[1], az = s1p[2];

  const float is3 = 0.5773502691896258f;  // 1/sqrt(3)
  const float is2 = 0.7071067811865476f;  // 1/sqrt(2)
  const float is6 = 0.4082482904638630f;  // 1/sqrt(6)

  const float t0e = (ax * bx + ay * by + az * bz) * is3;
  const float t1e0 = (ay * bz - az * by) * is2;
  const float t1e1 = (az * bx - ax * bz) * is2;
  const float t1e2 = (ax * by - ay * bx) * is2;
  const float t2e0 = (ax * by + ay * bx) * is2;
  const float t2e1 = (ay * bz + az * by) * is2;
  const float t2e2 = (2.0f * az * bz - ax * bx - ay * by) * is6;
  const float t2e3 = (ax * bz + az * bx) * is2;
  const float t2e4 = (ax * bx - ay * by) * is2;

  float* a0r = A0 + (size_t)r * 64;
  unsafeAtomicAdd(a0r + k, s0);
  unsafeAtomicAdd(a0r + 32 + k, t0e);

  const float sh[3]  = {bx, by, bz};
  const float s1v[3] = {ax, ay, az};
  const float t1e[3] = {t1e0, t1e1, t1e2};
#pragma unroll
  for (int i = 0; i < 3; ++i) {
    float* p = A1o + (size_t)i * N_NODES * 64 + (size_t)r * 64;
    unsafeAtomicAdd(p + k, s1v[i]);
    unsafeAtomicAdd(p + 32 + k, s0 * sh[i]);
    unsafeAtomicAdd(A1e + (size_t)i * N_NODES * 32 + (size_t)r * 32 + k, t1e[i]);
  }
  const float t2e[5] = {t2e0, t2e1, t2e2, t2e3, t2e4};
#pragma unroll
  for (int j = 0; j < 5; ++j)
    unsafeAtomicAdd(A2e + (size_t)j * N_NODES * 32 + (size_t)r * 32 + k, t2e[j]);
}

// ---------------------------------------------------------------------------
// Node kernel: one wave per 16-node tile; all GEMMs via bf16 WMMA with fp32
// accumulation. Intermediate h tiles bounce through a per-wave LDS 16x32
// buffer to convert C-fragment layout -> A-fragment layout. Weight B
// fragments are hoisted out of the component loops (loaded/converted once).
// ---------------------------------------------------------------------------
__global__ __launch_bounds__(128) void node_kernel(
    const float* __restrict__ x0, const float* __restrict__ x1,
    const float* __restrict__ A0, const float* __restrict__ A1o,
    const float* __restrict__ A1e, const float* __restrict__ A2e,
    const float* __restrict__ w_sc0, const float* __restrict__ w_sc1o,
    const float* __restrict__ w_pre0, const float* __restrict__ w_pre1o,
    const float* __restrict__ w_pre1e, const float* __restrict__ w_pre2e,
    const float* __restrict__ w_post0, const float* __restrict__ w_post1o,
    const float* __restrict__ w_post1e, const float* __restrict__ w_post2e,
    float* __restrict__ out) {
  __shared__ float hbuf_s[4][16 * 32];
  const int wave = threadIdx.x >> 5;
  const int lane = threadIdx.x & 31;
  const int tile = blockIdx.x * 4 + wave;
  if (tile * 16 >= N_NODES) return;   // wave-uniform: EXEC stays all-1s for WMMA
  const int row0 = tile * 16;
  float* hb = hbuf_s[wave];

  const float inv = 0.25f;                  // 1/sqrt(AVG_NEIGH)
  const float r8  = 0.125f;                 // 1/sqrt(64)
  const float r32 = 0.17677669529663687f;   // 1/sqrt(32)

  // ----- l=0: h0 = gelu(lin(a0, w_pre0)); y0 = lin(x0,w_sc0)+lin(h0,w_post0)
  {
#pragma unroll
    for (int nh = 0; nh < 2; ++nh) {
      v8f acc = {};
#pragma unroll
      for (int kc = 0; kc < 2; ++kc)
        acc = wmma_bf16(load_a_frag(A0, 64, 1, row0, kc * 32, lane),
                        load_b_frag(w_pre0, kc * 32, nh * 16, lane), acc);
#pragma unroll
      for (int v = 0; v < 8; ++v) acc[v] = gelu_tanh(acc[v] * inv * r8);
      store_c_lds(hb, acc, nh * 16, lane);
    }
#pragma unroll
    for (int nh = 0; nh < 2; ++nh) {
      v8f acc = {};
      acc = wmma_bf16(load_a_frag(x0, 32, 1, row0, 0, lane),
                      load_b_frag(w_sc0, 0, nh * 16, lane), acc);
      acc = wmma_bf16(load_a_frag(hb, 32, 1, 0, 0, lane),
                      load_b_frag(w_post0, 0, nh * 16, lane), acc);
#pragma unroll
      for (int v = 0; v < 8; ++v) acc[v] *= r32;
      store_c_out(out, row0, 0, 1, acc, nh * 16, lane);
    }
  }

  // ----- l=1 odd: h1o_i = lin(a1o_i,w_pre1o); y1o_i = lin(x1_i,w_sc1o)+lin(h1o_i,w_post1o)
  {
    v16bf b_pre[2][2], b_sc[2], b_post[2];   // hoisted: shared across i=0..2
#pragma unroll
    for (int nh = 0; nh < 2; ++nh) {
#pragma unroll
      for (int kc = 0; kc < 2; ++kc)
        b_pre[nh][kc] = load_b_frag(w_pre1o, kc * 32, nh * 16, lane);
      b_sc[nh]   = load_b_frag(w_sc1o, 0, nh * 16, lane);
      b_post[nh] = load_b_frag(w_post1o, 0, nh * 16, lane);
    }
    for (int i = 0; i < 3; ++i) {
      const float* A1oi = A1o + (size_t)i * N_NODES * 64;
#pragma unroll
      for (int nh = 0; nh < 2; ++nh) {
        v8f acc = {};
#pragma unroll
        for (int kc = 0; kc < 2; ++kc)
          acc = wmma_bf16(load_a_frag(A1oi, 64, 1, row0, kc * 32, lane),
                          b_pre[nh][kc], acc);
#pragma unroll
        for (int v = 0; v < 8; ++v) acc[v] *= inv * r8;
        store_c_lds(hb, acc, nh * 16, lane);
      }
#pragma unroll
      for (int nh = 0; nh < 2; ++nh) {
        v8f acc = {};
        acc = wmma_bf16(load_a_frag(x1 + i, 96, 3, row0, 0, lane), b_sc[nh], acc);
        acc = wmma_bf16(load_a_frag(hb, 32, 1, 0, 0, lane), b_post[nh], acc);
#pragma unroll
        for (int v = 0; v < 8; ++v) acc[v] *= r32;
        store_c_out(out, row0, 32 + i, 3, acc, nh * 16, lane);
      }
    }
  }

  // ----- l=1 even: y1e_i = lin(lin(a1e_i,w_pre1e), w_post1e)
  {
    v16bf b_pre[2], b_post[2];               // hoisted: shared across i=0..2
#pragma unroll
    for (int nh = 0; nh < 2; ++nh) {
      b_pre[nh]  = load_b_frag(w_pre1e, 0, nh * 16, lane);
      b_post[nh] = load_b_frag(w_post1e, 0, nh * 16, lane);
    }
    for (int i = 0; i < 3; ++i) {
      const float* A1ei = A1e + (size_t)i * N_NODES * 32;
#pragma unroll
      for (int nh = 0; nh < 2; ++nh) {
        v8f acc = {};
        acc = wmma_bf16(load_a_frag(A1ei, 32, 1, row0, 0, lane), b_pre[nh], acc);
#pragma unroll
        for (int v = 0; v < 8; ++v) acc[v] *= inv * r32;
        store_c_lds(hb, acc, nh * 16, lane);
      }
#pragma unroll
      for (int nh = 0; nh < 2; ++nh) {
        v8f acc = {};
        acc = wmma_bf16(load_a_frag(hb, 32, 1, 0, 0, lane), b_post[nh], acc);
#pragma unroll
        for (int v = 0; v < 8; ++v) acc[v] *= r32;
        store_c_out(out, row0, 128 + i, 3, acc, nh * 16, lane);
      }
    }
  }

  // ----- l=2 even: y2e_j = lin(lin(a2e_j,w_pre2e), w_post2e)
  {
    v16bf b_pre[2], b_post[2];               // hoisted: shared across j=0..4
#pragma unroll
    for (int nh = 0; nh < 2; ++nh) {
      b_pre[nh]  = load_b_frag(w_pre2e, 0, nh * 16, lane);
      b_post[nh] = load_b_frag(w_post2e, 0, nh * 16, lane);
    }
    for (int j = 0; j < 5; ++j) {
      const float* A2ej = A2e + (size_t)j * N_NODES * 32;
#pragma unroll
      for (int nh = 0; nh < 2; ++nh) {
        v8f acc = {};
        acc = wmma_bf16(load_a_frag(A2ej, 32, 1, row0, 0, lane), b_pre[nh], acc);
#pragma unroll
        for (int v = 0; v < 8; ++v) acc[v] *= inv * r32;
        store_c_lds(hb, acc, nh * 16, lane);
      }
#pragma unroll
      for (int nh = 0; nh < 2; ++nh) {
        v8f acc = {};
        acc = wmma_bf16(load_a_frag(hb, 32, 1, 0, 0, lane), b_post[nh], acc);
#pragma unroll
        for (int v = 0; v < 8; ++v) acc[v] *= r32;
        store_c_out(out, row0, 224 + j, 5, acc, nh * 16, lane);
      }
    }
  }
}

// ---------------------------------------------------------------------------
extern "C" void kernel_launch(void* const* d_in, const int* in_sizes, int n_in,
                              void* d_out, int out_size, void* d_ws, size_t ws_size,
                              hipStream_t stream) {
  const float* positions = (const float*)d_in[0];
  const float* x0        = (const float*)d_in[1];
  const float* x1        = (const float*)d_in[2];
  const int*   senders   = (const int*)d_in[3];
  const int*   receivers = (const int*)d_in[4];
  const float* w_sc0   = (const float*)d_in[5];
  const float* w_sc1o  = (const float*)d_in[6];
  const float* w_pre0  = (const float*)d_in[7];
  const float* w_pre1o = (const float*)d_in[8];
  const float* w_pre1e = (const float*)d_in[9];
  const float* w_pre2e = (const float*)d_in[10];
  const float* w_post0  = (const float*)d_in[11];
  const float* w_post1o = (const float*)d_in[12];
  const float* w_post1e = (const float*)d_in[13];
  const float* w_post2e = (const float*)d_in[14];
  float* out = (float*)d_out;

  // Workspace: fp32 segment-sum accumulators, 512 floats per node (~41 MB)
  float* A0  = (float*)d_ws;                   // N x 64
  float* A1o = A0 + (size_t)N_NODES * 64;      // 3 planes of N x 64
  float* A1e = A1o + (size_t)3 * N_NODES * 64; // 3 planes of N x 32
  float* A2e = A1e + (size_t)3 * N_NODES * 32; // 5 planes of N x 32
  const size_t acc_bytes = (size_t)N_NODES * 512 * sizeof(float);
  hipMemsetAsync(d_ws, 0, acc_bytes, stream);

  edge_kernel<<<N_EDGES / 8, 256, 0, stream>>>(positions, x0, x1, senders,
                                               receivers, A0, A1o, A1e, A2e);

  const int n_tiles = N_NODES / 16;  // 1250
  node_kernel<<<(n_tiles + 3) / 4, 128, 0, stream>>>(
      x0, x1, A0, A1o, A1e, A2e, w_sc0, w_sc1o, w_pre0, w_pre1o, w_pre1e,
      w_pre2e, w_post0, w_post1o, w_post1e, w_post2e, out);
}
